// SAGE_11871289606693
// MI455X (gfx1250) — matrix operand order repaired and checked by reading, not compiled, not run on hardware
//
#include <hip/hip_runtime.h>
#include <hip/hip_bf16.h>

// ---------------- problem constants ----------------
#define NN     50000      // nodes (3125 * 16 exactly -> no row remainder)
#define NE_    800000     // edges
#define DIN    128
#define DHID   256
#define DOUT   128
#define NEXP   8

typedef __attribute__((ext_vector_type(16))) __bf16 bf16x16;
typedef __attribute__((ext_vector_type(8)))  float  f32x8;

// WMMA A-fragment K-swizzle for 16-bit operands (16x32 tile, wave32):
// lane group 0 holds K={0..7, 16..23}, group 1 holds K={8..15, 24..31}.
// We store LDS rows so each lane's 16 values are contiguous (32B -> b128 loads).
__device__ __forceinline__ int swz(int k) {
    int kb = k >> 5, r = k & 31, lg, jj;
    if      (r < 8)  { lg = 0; jj = r;            }
    else if (r < 16) { lg = 1; jj = r - 8;        }
    else if (r < 24) { lg = 0; jj = 8 + (r - 16); }
    else             { lg = 1; jj = 8 + (r - 24); }
    return kb * 32 + lg * 16 + jj;
}

// ---------------- small utility kernels ----------------
__global__ void zero_f(float* __restrict__ p, int n) {
    int t = blockIdx.x * blockDim.x + threadIdx.x;
    if (t < n) p[t] = 0.0f;
}

__global__ void degree_kernel(const long long* __restrict__ dst, float* __restrict__ cnt, int e) {
    int t = blockIdx.x * blockDim.x + threadIdx.x;
    if (t < e) atomicAdd(&cnt[(int)dst[t]], 1.0f);
}

__global__ void recip_kernel(float* __restrict__ cnt, int n) {
    int t = blockIdx.x * blockDim.x + threadIdx.x;
    if (t < n) cnt[t] = 1.0f / fmaxf(cnt[t], 1.0f);
}

// scatter-add of x[src] rows into agg[dst] rows; one thread handles 4 floats
__global__ void scatter_add(const float* __restrict__ X, const long long* __restrict__ src,
                            const long long* __restrict__ dst, float* __restrict__ Agg,
                            int d, int e_total) {
    int per = d >> 2;
    int t = blockIdx.x * blockDim.x + threadIdx.x;
    if (t >= e_total * per) return;
    int e = t / per;
    int q = (t - e * per) << 2;
    int s = (int)src[e], dd = (int)dst[e];
    const float4 v = *(const float4*)(X + (size_t)s * d + q);
    float* b = Agg + (size_t)dd * d + q;
    atomicAdd(b + 0, v.x);
    atomicAdd(b + 1, v.y);
    atomicAdd(b + 2, v.z);
    atomicAdd(b + 3, v.w);
}

// Pack fp32 W[K][N] (row-major) -> bf16 WMMA-B layout:
// P[((kb*N + n)*2 + lg)*16 + j] = bf16( W[(kb*32 + (j<8 ? lg*8 + j : 16 + lg*8 + j-8))*N + n] )
// (stacked experts pack correctly because K is a multiple of 32)
__global__ void prepack(const float* __restrict__ W, __bf16* __restrict__ P, int Ktot, int N) {
    int t = blockIdx.x * blockDim.x + threadIdx.x;
    if (t >= Ktot * N) return;
    int j  = t & 15;
    int lg = (t >> 4) & 1;
    int n  = (t >> 5) % N;
    int kb = t / (32 * N);
    int k  = kb * 32 + (j < 8 ? lg * 8 + j : 16 + lg * 8 + (j - 8));
    P[t] = (__bf16)W[(size_t)k * N + n];
}

__global__ void finalize_kernel(const float* __restrict__ gsum, float* __restrict__ out) {
    out[0] = (gsum[0] + gsum[1]) * (0.5f / (float)NN);
}

// ---------------- fused layer kernel ----------------
// Per block: one 16-row node tile, 8 waves x 16 output columns (128 cols / block).
// MOE path: gate -> softmax/argmax/std -> masked per-expert WMMA accumulated into
// one C (rows are disjoint across experts) + root GEMM + bias + optional ReLU.
template <int K, int NOUT, bool MOE, bool RELU>
__global__ void __launch_bounds__(256)
layer_kernel(const float* __restrict__ Agg,   // [NN, K] un-normalized sums
             const float* __restrict__ Xin,   // [NN, K] root input
             const float* __restrict__ invd,  // [NN] 1/max(deg,1)
             const float* __restrict__ wg,    // [K, 8]   (MOE only)
             const float* __restrict__ bg,    // [8]      (MOE only)
             const __bf16* __restrict__ pWe,  // packed experts [8,K,NOUT] or single [K,NOUT]
             const float* __restrict__ be,    // [8, NOUT] (MOE only)
             const __bf16* __restrict__ pWr,  // packed root [K, NOUT]
             const float* __restrict__ bl,    // [NOUT]    (plain only)
             float* __restrict__ Out,         // [NN, NOUT]
             float* __restrict__ gsum)        // scalar accumulator (MOE only)
{
    __shared__ __bf16 sA[16 * K];     // normalized agg tile, WMMA-A swizzled
    __shared__ __bf16 sX[16 * K];     // root input tile,     WMMA-A swizzled
    __shared__ float  sLogit[16 * NEXP];
    __shared__ int    sIdx[16];
    __shared__ float  sStd[16];
    __shared__ int    sPresent;

    const int row0 = blockIdx.x * 16;

    // ---- Phase A: stage tiles into LDS (fp32 -> bf16, swizzled, mean-normalize A)
    for (int i = threadIdx.x; i < 16 * (K / 4); i += 256) {
        int m  = i / (K / 4);
        int kq = (i - m * (K / 4)) << 2;           // multiple of 4, stays in one octet
        float sc = invd[row0 + m];
        const float4 av = *(const float4*)(Agg + (size_t)(row0 + m) * K + kq);
        const float4 xv = *(const float4*)(Xin + (size_t)(row0 + m) * K + kq);
        int d0 = m * K + swz(kq);                  // swz(kq+t) == swz(kq)+t for t<4
        sA[d0 + 0] = (__bf16)(av.x * sc);
        sA[d0 + 1] = (__bf16)(av.y * sc);
        sA[d0 + 2] = (__bf16)(av.z * sc);
        sA[d0 + 3] = (__bf16)(av.w * sc);
        sX[d0 + 0] = (__bf16)xv.x;
        sX[d0 + 1] = (__bf16)xv.y;
        sX[d0 + 2] = (__bf16)xv.z;
        sX[d0 + 3] = (__bf16)xv.w;
    }
    __syncthreads();

    // ---- Phase B: gate logits (16 rows x 8 experts), one per thread
    if (MOE) {
        if (threadIdx.x < 16 * NEXP) {
            int r = threadIdx.x & 15;
            int e = threadIdx.x >> 4;
            float acc = bg[e];
            for (int k = 0; k < K; ++k)
                acc += (float)sA[r * K + swz(k)] * wg[k * NEXP + e];
            sLogit[r * NEXP + e] = acc;
        }
        __syncthreads();

        // ---- Phase C: softmax / argmax / unbiased std per row
        if (threadIdx.x < 16) {
            int r = threadIdx.x;
            float l[NEXP];
            float mx = -1e30f;
            int am = 0;
            for (int e = 0; e < NEXP; ++e) {
                l[e] = sLogit[r * NEXP + e];
                if (l[e] > mx) { mx = l[e]; am = e; }   // first max, like argmax
            }
            float se = 0.0f;
            for (int e = 0; e < NEXP; ++e) { l[e] = __expf(l[e] - mx); se += l[e]; }
            float is = 1.0f / se;
            float mean = 0.0f;
            for (int e = 0; e < NEXP; ++e) { l[e] *= is; mean += l[e]; }
            mean *= (1.0f / NEXP);
            float var = 0.0f;
            for (int e = 0; e < NEXP; ++e) { float d = l[e] - mean; var += d * d; }
            var *= (1.0f / (NEXP - 1));
            sIdx[r] = am;
            sStd[r] = sqrtf(var);
        }
        __syncthreads();
        if (threadIdx.x == 0) {
            float s = 0.0f;
            int pm = 0;
            for (int r = 0; r < 16; ++r) { s += sStd[r]; pm |= 1 << sIdx[r]; }
            atomicAdd(gsum, s);
            sPresent = pm;
        }
        __syncthreads();
    }

    // ---- Phase D: WMMA GEMMs. Wave w owns 16 output columns.
    const int lane = threadIdx.x & 31;
    const int w    = threadIdx.x >> 5;
    const int lg   = lane >> 4;       // K-half selector
    const int mr   = lane & 15;       // A row / B column within tile
    const int nb   = blockIdx.y * 128 + w * 16;
    const int col  = nb + mr;
    constexpr int KB = K / 32;

    f32x8 acc = {};
    const bf16x16 z = {};

    if (MOE) {
        const int myExp = sIdx[mr];
        const int pm = sPresent;      // wave-uniform -> EXEC stays all-ones
        for (int e = 0; e < NEXP; ++e) {
            if (!((pm >> e) & 1)) continue;
            const bool sel = (myExp == e);
            for (int kb = 0; kb < KB; ++kb) {
                bf16x16 a = *(const bf16x16*)(sA + mr * K + kb * 32 + lg * 16);
                a = sel ? a : z;
                bf16x16 b = *(const bf16x16*)(pWe + (((size_t)(e * KB + kb)) * NOUT + col) * 32 + lg * 16);
                acc = __builtin_amdgcn_wmma_f32_16x16x32_bf16(false, a, false, b, (short)0, acc, false, false);
            }
        }
    } else {
        for (int kb = 0; kb < KB; ++kb) {
            bf16x16 a = *(const bf16x16*)(sA + mr * K + kb * 32 + lg * 16);
            bf16x16 b = *(const bf16x16*)(pWe + (((size_t)kb) * NOUT + col) * 32 + lg * 16);
            acc = __builtin_amdgcn_wmma_f32_16x16x32_bf16(false, a, false, b, (short)0, acc, false, false);
        }
    }
    // root GEMM: x @ Wr
    for (int kb = 0; kb < KB; ++kb) {
        bf16x16 a = *(const bf16x16*)(sX + mr * K + kb * 32 + lg * 16);
        bf16x16 b = *(const bf16x16*)(pWr + (((size_t)kb) * NOUT + col) * 32 + lg * 16);
        acc = __builtin_amdgcn_wmma_f32_16x16x32_bf16(false, a, false, b, (short)0, acc, false, false);
    }

    // ---- Epilogue: bias (+expert bias), ReLU, store.
    // C layout: element r -> row (lg*8 + r), column = lane%16.
#pragma unroll
    for (int r = 0; r < 8; ++r) {
        int mrow = lg * 8 + r;
        float v = acc[r];
        if (MOE) v += be[(size_t)sIdx[mrow] * NOUT + col];
        else     v += bl[col];
        if (RELU) v = fmaxf(v, 0.0f);
        Out[(size_t)(row0 + mrow) * NOUT + col] = v;
    }
}

// ---------------- host-side orchestration ----------------
extern "C" void kernel_launch(void* const* d_in, const int* in_sizes, int n_in,
                              void* d_out, int out_size, void* d_ws, size_t ws_size,
                              hipStream_t stream) {
    (void)in_sizes; (void)n_in; (void)out_size; (void)ws_size;

    const float*     x   = (const float*)d_in[0];
    const long long* ei  = (const long long*)d_in[1];   // [2, E] int64
    const float*     wg1 = (const float*)d_in[2];
    const float*     bg1 = (const float*)d_in[3];
    const float*     we1 = (const float*)d_in[4];
    const float*     be1 = (const float*)d_in[5];
    const float*     wr1 = (const float*)d_in[6];
    const float*     wg2 = (const float*)d_in[7];
    const float*     bg2 = (const float*)d_in[8];
    const float*     we2 = (const float*)d_in[9];
    const float*     be2 = (const float*)d_in[10];
    const float*     wr2 = (const float*)d_in[11];
    const float*     wl3 = (const float*)d_in[12];
    const float*     bl3 = (const float*)d_in[13];
    const float*     wr3 = (const float*)d_in[14];

    const long long* src = ei;
    const long long* dst = ei + NE_;

    // workspace arenas (fp32 unless noted); everything re-initialized per call
    float* buf0 = (float*)d_ws;                       // NN*256 : agg1 (128 used), later h2
    float* buf1 = buf0 + (size_t)NN * 256;            // NN*256 : h1
    float* buf2 = buf1 + (size_t)NN * 256;            // NN*256 : agg2, then agg3
    float* cnt  = buf2 + (size_t)NN * 256;            // NN     : degree -> reciprocal
    float* gsum = cnt + NN;                           // 2 scalars (+pad to 16)
    __bf16* pwe1 = (__bf16*)(gsum + 16);              // 8*128*256
    __bf16* pwr1 = pwe1 + 8 * 128 * 256;              // 128*256
    __bf16* pwe2 = pwr1 + 128 * 256;                  // 8*256*256
    __bf16* pwr2 = pwe2 + 8 * 256 * 256;              // 256*256
    __bf16* pwl3 = pwr2 + 256 * 256;                  // 256*128
    __bf16* pwr3 = pwl3 + 256 * 128;                  // 256*128

    const int B = 256;
    auto nb = [](int n) { return (n + 255) / 256; };

    // init counters + pack weights to bf16 WMMA-B layout
    zero_f<<<nb(NN + 16), B, 0, stream>>>(cnt, NN + 16);
    prepack<<<nb(8 * 128 * 256), B, 0, stream>>>(we1, pwe1, 8 * 128, 256);
    prepack<<<nb(128 * 256),     B, 0, stream>>>(wr1, pwr1, 128, 256);
    prepack<<<nb(8 * 256 * 256), B, 0, stream>>>(we2, pwe2, 8 * 256, 256);
    prepack<<<nb(256 * 256),     B, 0, stream>>>(wr2, pwr2, 256, 256);
    prepack<<<nb(256 * 128),     B, 0, stream>>>(wl3, pwl3, 256, 128);
    prepack<<<nb(256 * 128),     B, 0, stream>>>(wr3, pwr3, 256, 128);

    // degree -> 1/max(deg,1)
    degree_kernel<<<nb(NE_), B, 0, stream>>>(dst, cnt, NE_);
    recip_kernel<<<nb(NN), B, 0, stream>>>(cnt, NN);

    // ---- layer 1: agg(x) -> MoE -> relu -> h1 (buf1)
    zero_f<<<nb(NN * 128), B, 0, stream>>>(buf0, NN * 128);
    scatter_add<<<nb(NE_ * (128 / 4)), B, 0, stream>>>(x, src, dst, buf0, 128, NE_);
    layer_kernel<128, 256, true, true><<<dim3(NN / 16, 2), B, 0, stream>>>(
        buf0, x, cnt, wg1, bg1, pwe1, be1, pwr1, nullptr, buf1, gsum);

    // ---- layer 2: agg(h1) -> MoE -> relu -> h2 (buf0)
    zero_f<<<nb(NN * 256), B, 0, stream>>>(buf2, NN * 256);
    scatter_add<<<nb(NE_ * (256 / 4)), B, 0, stream>>>(buf1, src, dst, buf2, 256, NE_);
    layer_kernel<256, 256, true, true><<<dim3(NN / 16, 2), B, 0, stream>>>(
        buf2, buf1, cnt, wg2, bg2, pwe2, be2, pwr2, nullptr, buf0, gsum + 1);

    // ---- layer 3: agg(h2) -> plain SAGE -> d_out
    zero_f<<<nb(NN * 256), B, 0, stream>>>(buf2, NN * 256);
    scatter_add<<<nb(NE_ * (256 / 4)), B, 0, stream>>>(buf0, src, dst, buf2, 256, NE_);
    layer_kernel<256, 128, false, false><<<dim3(NN / 16, 1), B, 0, stream>>>(
        buf2, buf0, cnt, nullptr, nullptr, pwl3, nullptr, pwr3, bl3, (float*)d_out, nullptr);

    // ---- gate_std_mean scalar
    finalize_kernel<<<1, 1, 0, stream>>>(gsum, (float*)d_out + (size_t)NN * DOUT);
}
